// ZetaModel_16973710753851
// MI455X (gfx1250) — matrix operand order repaired and checked by reading, not compiled, not run on hardware
//
#include <hip/hip_runtime.h>
#include <math.h>

typedef __attribute__((ext_vector_type(16))) _Float16 v16h;
typedef __attribute__((ext_vector_type(8)))  float    v8f;

#define BATCH 8
#define SEQL  2048
#define TOK   (BATCH * SEQL)   // 16384
#define DM    320
#define DI    1280
#define NST   16
#define DCONV 8
#define DTRANK 20
#define NBLK  4
#define DBCW  (DTRANK + 2 * NST)  // 52
#define CHK   64                  // scan/conv chunk length
#define NCHK  (SEQL / CHK)        // 32 chunks per sequence

// ---------------------------------------------------------------------------
// start_max = max over x[:,:,2]
// ---------------------------------------------------------------------------
__global__ __launch_bounds__(1024) void max_kernel(const float* __restrict__ x,
                                                   float* __restrict__ smax, int TL) {
  __shared__ float red[1024];
  float m = 0.0f;
  for (int i = threadIdx.x; i < TL; i += 1024) m = fmaxf(m, x[(size_t)i * 4 + 2]);
  red[threadIdx.x] = m;
  __syncthreads();
  for (int s = 512; s > 0; s >>= 1) {
    if (threadIdx.x < (unsigned)s) red[threadIdx.x] = fmaxf(red[threadIdx.x], red[threadIdx.x + s]);
    __syncthreads();
  }
  if (threadIdx.x == 0) smax[0] = red[0];
}

// ---------------------------------------------------------------------------
// f32 -> f16 weight conversion (once per launch)
// ---------------------------------------------------------------------------
__global__ __launch_bounds__(256) void cvt_kernel(const float* __restrict__ src,
                                                  _Float16* __restrict__ dst, int n) {
  int i = blockIdx.x * 256 + threadIdx.x;
  if (i < n) dst[i] = (_Float16)src[i];
}

// ---------------------------------------------------------------------------
// input normalization + fc (4 -> 320)
// ---------------------------------------------------------------------------
__global__ __launch_bounds__(256) void embed_kernel(const float* __restrict__ x,
                                                    const float* __restrict__ fw,
                                                    const float* __restrict__ fb,
                                                    const float* __restrict__ smax,
                                                    _Float16* __restrict__ act) {
  size_t gid = (size_t)blockIdx.x * 256 + threadIdx.x;  // over TOK*DM
  int tok = (int)(gid / DM), c = (int)(gid % DM);
  const float* xr = x + (size_t)tok * 4;
  float inv = 1.0f / smax[0];
  float x0 = xr[0] * (1.0f / 255.0f);
  float x1 = xr[1] * (1.0f / 255.0f);
  float x2 = xr[2] * inv;
  float x3 = xr[3];
  const float* w = fw + (size_t)c * 4;
  float v = fb[c] + x0 * w[0] + x1 * w[1] + x2 * w[2] + x3 * w[3];
  act[gid] = (_Float16)v;
}

// ---------------------------------------------------------------------------
// WMMA GEMM:  C[M,N] = act( A[M,K] @ W[N,K]^T + bias )     (all f16, f32 acc)
// Block tile 128x64, 8 waves, each wave 2x2 WMMA 16x16 tiles, K-step 32.
// act: 0 = none, 1 = tanh, 2 = elu
// ---------------------------------------------------------------------------
#define BM 128
#define BN 64
#define BK 32
#define LDP (BK + 8)

__global__ __launch_bounds__(256) void gemm_f16_wmma(const _Float16* __restrict__ A,
                                                     const _Float16* __restrict__ W,
                                                     const float* __restrict__ bias,
                                                     _Float16* __restrict__ C,
                                                     int M, int N, int K, int act) {
  __shared__ _Float16 As[BM][LDP];
  __shared__ _Float16 Bs[BN][LDP];

  const int tid  = threadIdx.x;
  const int lane = tid & 31;
  const int wave = tid >> 5;
  const int wm   = (wave & 3) * 32;
  const int wn   = (wave >> 2) * 32;
  const int mBase = blockIdx.y * BM;
  const int nBase = blockIdx.x * BN;

  const int lrow  = lane & 15;
  const int khalf = lane >> 4;   // 0: first K-half pattern, 1: second

  v8f acc[2][2] = {};

  for (int kk = 0; kk < K; kk += BK) {
    // --- A tile: 128x32 halfs = 512 chunks of 8 halfs (16B), 2 per thread
#pragma unroll
    for (int t = 0; t < 2; ++t) {
      int c  = tid + t * 256;
      int r  = c >> 2;
      int cl = (c & 3) * 8;
      uint4 v = *reinterpret_cast<const uint4*>(&A[(size_t)(mBase + r) * K + kk + cl]);
      *reinterpret_cast<uint4*>(&As[r][cl]) = v;
    }
    // --- B tile: 64x32 halfs = 256 chunks, 1 per thread
    {
      int r  = tid >> 2;
      int cl = (tid & 3) * 8;
      uint4 v = *reinterpret_cast<const uint4*>(&W[(size_t)(nBase + r) * K + kk + cl]);
      *reinterpret_cast<uint4*>(&Bs[r][cl]) = v;
    }
    if (kk + BK < K) {
      __builtin_prefetch(&A[(size_t)(mBase + (tid >> 1)) * K + kk + BK], 0, 3);
      __builtin_prefetch(&W[(size_t)(nBase + (tid & 63)) * K + kk + BK], 0, 3);
    }
    __syncthreads();

    // --- fragments (ISA 7.12.2 16-bit A 16x32 / B 32x16 layouts)
    v16h afrag[2], bfrag[2];
#pragma unroll
    for (int i = 0; i < 2; ++i) {
      int m = wm + i * 16 + lrow;
      uint4* u = reinterpret_cast<uint4*>(&afrag[i]);
      u[0] = *reinterpret_cast<const uint4*>(&As[m][khalf * 8]);
      u[1] = *reinterpret_cast<const uint4*>(&As[m][16 + khalf * 8]);
    }
#pragma unroll
    for (int j = 0; j < 2; ++j) {
      int n = wn + j * 16 + lrow;
      uint4* u = reinterpret_cast<uint4*>(&bfrag[j]);
      u[0] = *reinterpret_cast<const uint4*>(&Bs[n][khalf * 8]);
      u[1] = *reinterpret_cast<const uint4*>(&Bs[n][16 + khalf * 8]);
    }
#pragma unroll
    for (int i = 0; i < 2; ++i)
#pragma unroll
      for (int j = 0; j < 2; ++j)
        acc[i][j] = __builtin_amdgcn_wmma_f32_16x16x32_f16(
            false, afrag[i], false, bfrag[j], (short)0, acc[i][j], false, false);
    __syncthreads();
  }

  // --- epilogue: C/D layout VGPR v -> M = v (+8 for lanes 16..31), N = lane&15
#pragma unroll
  for (int i = 0; i < 2; ++i) {
#pragma unroll
    for (int j = 0; j < 2; ++j) {
      int ncol = nBase + wn + j * 16 + (lane & 15);
      float bv = bias ? bias[ncol] : 0.0f;
#pragma unroll
      for (int v = 0; v < 8; ++v) {
        int mrow = mBase + wm + i * 16 + v + ((lane >= 16) ? 8 : 0);
        float r = acc[i][j][v] + bv;
        if (act == 1)      r = tanhf(r);
        else if (act == 2) r = (r > 0.0f) ? r : (__expf(r) - 1.0f);
        C[(size_t)mrow * N + ncol] = (_Float16)r;
      }
    }
  }
}

// ---------------------------------------------------------------------------
// depthwise causal conv (kernel 8) + bias + SiLU, chunk-parallel over L
// (no recurrence: each chunk of 64 steps just preloads a 7-tap halo)
// grid (BATCH, DI/256, NCHK)
// ---------------------------------------------------------------------------
__global__ __launch_bounds__(256) void conv_silu_kernel(const _Float16* __restrict__ xin,
                                                        _Float16* __restrict__ xout,
                                                        const float* __restrict__ cw,
                                                        const float* __restrict__ cb) {
  int b = blockIdx.x;
  int d = blockIdx.y * 256 + threadIdx.x;
  int c = blockIdx.z;
  float w[DCONV];
#pragma unroll
  for (int k = 0; k < DCONV; ++k) w[k] = cw[(size_t)d * DCONV + k];
  float bias = cb[d];
  const int tstart = c * CHK;
  float win[DCONV] = {0, 0, 0, 0, 0, 0, 0, 0};
  // halo: win[1..7] = x[tstart-7 .. tstart-1] (zero-padded)
#pragma unroll
  for (int k = 0; k < DCONV - 1; ++k) {
    int tt = tstart - (DCONV - 1) + k;
    win[k + 1] = (tt >= 0) ? (float)xin[((size_t)b * SEQL + tt) * DI + d] : 0.0f;
  }
  size_t idx = ((size_t)b * SEQL + tstart) * DI + d;
  for (int t = 0; t < CHK; ++t, idx += DI) {
#pragma unroll
    for (int k = 0; k < DCONV - 1; ++k) win[k] = win[k + 1];
    win[DCONV - 1] = (float)xin[idx];
    float a = bias;
#pragma unroll
    for (int k = 0; k < DCONV; ++k) a += w[k] * win[k];
    float s = a / (1.0f + __expf(-a));
    xout[idx] = (_Float16)s;
  }
}

// ---------------------------------------------------------------------------
// xproj: dbc[T,52] = xc[T,1280] @ W[52,1280]^T      (64-lane group per token)
// ---------------------------------------------------------------------------
__global__ __launch_bounds__(256) void xproj_kernel(const _Float16* __restrict__ xc,
                                                    const float* __restrict__ W,
                                                    float* __restrict__ dbc) {
  int tok = blockIdx.x * 4 + (threadIdx.x >> 6);
  int j   = threadIdx.x & 63;
  if (j >= DBCW) return;
  const _Float16* xr = xc + (size_t)tok * DI;
  const float*    wr = W + (size_t)j * DI;
  float acc = 0.0f;
  for (int k = 0; k < DI; k += 4) {
    acc += (float)xr[k] * wr[k] + (float)xr[k + 1] * wr[k + 1] +
           (float)xr[k + 2] * wr[k + 2] + (float)xr[k + 3] * wr[k + 3];
  }
  dbc[(size_t)tok * DBCW + j] = acc;
}

// ---------------------------------------------------------------------------
// dtproj + softplus: delta[T,DI] = softplus(dt[T,20] @ W[DI,20]^T + b)
// ---------------------------------------------------------------------------
__global__ __launch_bounds__(256) void dtproj_kernel(const float* __restrict__ dbc,
                                                     const float* __restrict__ W,
                                                     const float* __restrict__ bias,
                                                     _Float16* __restrict__ delta) {
  size_t gid = (size_t)blockIdx.x * 256 + threadIdx.x;  // over TOK*DI
  int tok = (int)(gid / DI);
  int d   = (int)(gid % DI);
  const float* dtv = dbc + (size_t)tok * DBCW;
  const float* wr  = W + (size_t)d * DTRANK;
  float acc = bias[d];
#pragma unroll
  for (int k = 0; k < DTRANK; ++k) acc += dtv[k] * wr[k];
  float sp = (acc > 20.0f) ? acc : log1pf(__expf(acc));
  delta[gid] = (_Float16)sp;
}

// ---------------------------------------------------------------------------
// chunked parallel scan. Recurrence h = dA*h + dx*B is linear diagonal, so
// each 64-step chunk reduces to an affine map (P, Q) per (b,d,n):
//   pass1:   P = prod(dA),  Q = sum of inputs propagated to chunk end
//   combine: sequentially fold 32 chunk maps -> incoming state per chunk
//   pass2:   re-run chunks in parallel from correct incoming state, emit y
// P/Q/Hin layout: [b][chunk][n][d] (d fastest -> coalesced)
// ---------------------------------------------------------------------------
__global__ __launch_bounds__(256) void scan_pass1(const _Float16* __restrict__ xc,
                                                  const _Float16* __restrict__ delta,
                                                  const float* __restrict__ dbc,
                                                  const float* __restrict__ A_log,
                                                  float* __restrict__ Pbuf,
                                                  float* __restrict__ Qbuf) {
  int b = blockIdx.x;
  int d = blockIdx.y * 256 + threadIdx.x;
  int c = blockIdx.z;
  float A[NST], P[NST], Q[NST];
#pragma unroll
  for (int n = 0; n < NST; ++n) {
    A[n] = -__expf(A_log[(size_t)d * NST + n]);
    P[n] = 1.0f;
    Q[n] = 0.0f;
  }
  size_t t0  = (size_t)b * SEQL + (size_t)c * CHK;
  size_t idx = t0 * DI + d;
  size_t row = t0 * DBCW;
  for (int t = 0; t < CHK; ++t, idx += DI, row += DBCW) {
    float dt = (float)delta[idx];
    float xv = (float)xc[idx];
    float dx = dt * xv;
    const float* Bp = dbc + row + DTRANK;
#pragma unroll
    for (int n = 0; n < NST; ++n) {
      float dA = __expf(dt * A[n]);
      P[n] *= dA;
      Q[n] = dA * Q[n] + dx * Bp[n];
    }
  }
  size_t o = (((size_t)b * NCHK + c) * NST) * DI + d;
#pragma unroll
  for (int n = 0; n < NST; ++n) {
    Pbuf[o + (size_t)n * DI] = P[n];
    Qbuf[o + (size_t)n * DI] = Q[n];
  }
}

__global__ __launch_bounds__(256) void scan_combine(float* __restrict__ Pbuf,
                                                    const float* __restrict__ Qbuf) {
  // grid (BATCH, DI/256); fold chunk maps, write incoming state over Pbuf
  int b = blockIdx.x;
  int d = blockIdx.y * 256 + threadIdx.x;
  float H[NST] = {};
  for (int c = 0; c < NCHK; ++c) {
    size_t o = (((size_t)b * NCHK + c) * NST) * DI + d;
#pragma unroll
    for (int n = 0; n < NST; ++n) {
      float Pv = Pbuf[o + (size_t)n * DI];
      float Qv = Qbuf[o + (size_t)n * DI];
      Pbuf[o + (size_t)n * DI] = H[n];   // incoming state for this chunk
      H[n] = Pv * H[n] + Qv;
    }
  }
}

__global__ __launch_bounds__(256) void scan_pass2(const _Float16* __restrict__ xc,
                                                  _Float16* __restrict__ zy,
                                                  const _Float16* __restrict__ delta,
                                                  const float* __restrict__ dbc,
                                                  const float* __restrict__ A_log,
                                                  const float* __restrict__ Dp,
                                                  const float* __restrict__ Hin) {
  int b = blockIdx.x;
  int d = blockIdx.y * 256 + threadIdx.x;
  int c = blockIdx.z;
  float A[NST], h[NST];
#pragma unroll
  for (int n = 0; n < NST; ++n) A[n] = -__expf(A_log[(size_t)d * NST + n]);
  float Dv = Dp[d];
  size_t o = (((size_t)b * NCHK + c) * NST) * DI + d;
#pragma unroll
  for (int n = 0; n < NST; ++n) h[n] = Hin[o + (size_t)n * DI];
  size_t t0  = (size_t)b * SEQL + (size_t)c * CHK;
  size_t idx = t0 * DI + d;
  size_t row = t0 * DBCW;
  for (int t = 0; t < CHK; ++t, idx += DI, row += DBCW) {
    float dt = (float)delta[idx];
    float xv = (float)xc[idx];
    float zv = (float)zy[idx];
    const float* Bp = dbc + row + DTRANK;
    const float* Cp = dbc + row + DTRANK + NST;
    float dx = dt * xv;
    float y = 0.0f;
#pragma unroll
    for (int n = 0; n < NST; ++n) {
      float dA = __expf(dt * A[n]);
      h[n] = dA * h[n] + dx * Bp[n];
      y += h[n] * Cp[n];
    }
    y += xv * Dv;
    y *= zv / (1.0f + __expf(-zv));
    zy[idx] = (_Float16)y;
  }
}

// ---------------------------------------------------------------------------
// head MLP on last token: 320 -> 512 -> 512 -> 1, scale, relu
// ---------------------------------------------------------------------------
__global__ __launch_bounds__(512) void head_kernel(const _Float16* __restrict__ act,
                                                   const float* __restrict__ w1, const float* __restrict__ b1,
                                                   const float* __restrict__ w2, const float* __restrict__ b2,
                                                   const float* __restrict__ w3, const float* __restrict__ b3,
                                                   const float* __restrict__ smax,
                                                   float* __restrict__ out) {
  __shared__ float hin[DM];
  __shared__ float h1[512];
  __shared__ float h2[512];
  int tid = threadIdx.x;
  for (int b = 0; b < BATCH; ++b) {
    if (tid < DM) hin[tid] = (float)act[((size_t)b * SEQL + (SEQL - 1)) * DM + tid];
    __syncthreads();
    float a = b1[tid];
    const float* wr = w1 + (size_t)tid * DM;
    for (int k = 0; k < DM; ++k) a += hin[k] * wr[k];
    h1[tid] = fmaxf(a, 0.0f);
    __syncthreads();
    a = b2[tid];
    wr = w2 + (size_t)tid * 512;
    for (int k = 0; k < 512; ++k) a += h1[k] * wr[k];
    h2[tid] = fmaxf(a, 0.0f);
    __syncthreads();
    h1[tid] = h2[tid] * w3[tid];
    __syncthreads();
    for (int s = 256; s > 0; s >>= 1) {
      if (tid < s) h1[tid] += h1[tid + s];
      __syncthreads();
    }
    if (tid == 0) {
      float v = h1[0] + b3[0];
      float scale = smax[0] / (1.0f + logf(2048.0f));
      out[b] = fmaxf(v * scale, 0.0f);
    }
    __syncthreads();
  }
}

// ---------------------------------------------------------------------------
// launch
// ---------------------------------------------------------------------------
extern "C" void kernel_launch(void* const* d_in, const int* in_sizes, int n_in,
                              void* d_out, int out_size, void* d_ws, size_t ws_size,
                              hipStream_t stream) {
  const float* x            = (const float*)d_in[0];
  const float* fc_w         = (const float*)d_in[1];
  const float* fc_b         = (const float*)d_in[2];
  const float* blk_lin_w    = (const float*)d_in[3];
  const float* blk_lin_b    = (const float*)d_in[4];
  const float* blk_inproj_w = (const float*)d_in[5];
  const float* blk_conv_w   = (const float*)d_in[6];
  const float* blk_conv_b   = (const float*)d_in[7];
  const float* blk_xproj_w  = (const float*)d_in[8];
  const float* blk_dtproj_w = (const float*)d_in[9];
  const float* blk_dtproj_b = (const float*)d_in[10];
  const float* blk_A_log    = (const float*)d_in[11];
  const float* blk_D        = (const float*)d_in[12];
  const float* blk_outproj_w= (const float*)d_in[13];
  const float* s1_w1        = (const float*)d_in[14];
  const float* s1_b1        = (const float*)d_in[15];
  const float* s1_w2        = (const float*)d_in[16];
  const float* s1_b2        = (const float*)d_in[17];
  const float* s1_w3        = (const float*)d_in[18];
  const float* s1_b3        = (const float*)d_in[19];
  float* out = (float*)d_out;

  // ---- workspace carve-up
  char* p = (char*)d_ws;
  auto carve = [&](size_t bytes) -> void* {
    void* r = (void*)p;
    p += (bytes + 255) & ~(size_t)255;
    return r;
  };
  const size_t nLin  = (size_t)NBLK * DM * DM;           // 409,600
  const size_t nInp  = (size_t)NBLK * 2 * DI * DM;       // 3,276,800
  const size_t nOutp = (size_t)NBLK * DM * DI;           // 1,638,400
  const size_t nPQ   = (size_t)BATCH * NCHK * NST * DI;  // 5,242,880

  float*    smax    = (float*)carve(sizeof(float));
  _Float16* lin16   = (_Float16*)carve(nLin  * 2);
  _Float16* inp16   = (_Float16*)carve(nInp  * 2);
  _Float16* outp16  = (_Float16*)carve(nOutp * 2);
  _Float16* actA    = (_Float16*)carve((size_t)TOK * DM * 2);
  _Float16* actB    = (_Float16*)carve((size_t)TOK * DM * 2);
  _Float16* xbuf    = (_Float16*)carve((size_t)TOK * DI * 2);
  _Float16* xcbuf   = (_Float16*)carve((size_t)TOK * DI * 2);
  _Float16* zbuf    = (_Float16*)carve((size_t)TOK * DI * 2);
  _Float16* delta16 = (_Float16*)carve((size_t)TOK * DI * 2);
  float*    dbc     = (float*)carve((size_t)TOK * DBCW * 4);
  float*    Pbuf    = (float*)carve(nPQ * 4);
  float*    Qbuf    = (float*)carve(nPQ * 4);

  // ---- prologue
  max_kernel<<<1, 1024, 0, stream>>>(x, smax, TOK);
  cvt_kernel<<<(int)((nLin  + 255) / 256), 256, 0, stream>>>(blk_lin_w,     lin16,  (int)nLin);
  cvt_kernel<<<(int)((nInp  + 255) / 256), 256, 0, stream>>>(blk_inproj_w,  inp16,  (int)nInp);
  cvt_kernel<<<(int)((nOutp + 255) / 256), 256, 0, stream>>>(blk_outproj_w, outp16, (int)nOutp);
  embed_kernel<<<(TOK * DM) / 256, 256, 0, stream>>>(x, fc_w, fc_b, smax, actA);

  // ---- mamba blocks
  for (int i = 0; i < NBLK; ++i) {
    // lin + tanh: [T,320] @ [320,320]^T
    gemm_f16_wmma<<<dim3(DM / BN, TOK / BM), 256, 0, stream>>>(
        actA, lin16 + (size_t)i * DM * DM, blk_lin_b + (size_t)i * DM, actB,
        TOK, DM, DM, 1);
    // inproj x-half and z-half: [T,320] @ [1280,320]^T
    gemm_f16_wmma<<<dim3(DI / BN, TOK / BM), 256, 0, stream>>>(
        actB, inp16 + (size_t)i * 2 * DI * DM, nullptr, xbuf, TOK, DI, DM, 0);
    gemm_f16_wmma<<<dim3(DI / BN, TOK / BM), 256, 0, stream>>>(
        actB, inp16 + (size_t)i * 2 * DI * DM + (size_t)DI * DM, nullptr, zbuf,
        TOK, DI, DM, 0);
    // depthwise causal conv + SiLU, chunk-parallel
    conv_silu_kernel<<<dim3(BATCH, DI / 256, NCHK), 256, 0, stream>>>(
        xbuf, xcbuf, blk_conv_w + (size_t)i * DI * DCONV, blk_conv_b + (size_t)i * DI);
    // xproj -> dbc
    xproj_kernel<<<TOK / 4, 256, 0, stream>>>(
        xcbuf, blk_xproj_w + (size_t)i * DBCW * DI, dbc);
    // dtproj + softplus -> delta
    dtproj_kernel<<<(TOK * DI) / 256, 256, 0, stream>>>(
        dbc, blk_dtproj_w + (size_t)i * DI * DTRANK, blk_dtproj_b + (size_t)i * DI, delta16);
    // chunked parallel scan (+ D skip + z gate); y overwrites zbuf
    scan_pass1<<<dim3(BATCH, DI / 256, NCHK), 256, 0, stream>>>(
        xcbuf, delta16, dbc, blk_A_log + (size_t)i * DI * NST, Pbuf, Qbuf);
    scan_combine<<<dim3(BATCH, DI / 256), 256, 0, stream>>>(Pbuf, Qbuf);
    scan_pass2<<<dim3(BATCH, DI / 256, NCHK), 256, 0, stream>>>(
        xcbuf, zbuf, delta16, dbc, blk_A_log + (size_t)i * DI * NST,
        blk_D + (size_t)i * DI, Pbuf);
    // outproj + elu: [T,1280] @ [320,1280]^T -> actA (next block input)
    gemm_f16_wmma<<<dim3(DM / BN, TOK / BM), 256, 0, stream>>>(
        zbuf, outp16 + (size_t)i * DM * DI, nullptr, actA, TOK, DM, DI, 2);
  }

  // ---- head
  head_kernel<<<1, 512, 0, stream>>>(actA, s1_w1, s1_b1, s1_w2, s1_b2, s1_w3, s1_b3,
                                     smax, out);
  (void)in_sizes; (void)n_in; (void)out_size; (void)ws_size;
}
